// BigramLanguageModel_23381801960051
// MI455X (gfx1250) — matrix operand order repaired and checked by reading, not compiled, not run on hardware
//
#include <hip/hip_runtime.h>

#define VOCAB  50257
#define EMBED  256
#define NHEAD  8
#define HEAD   32
#define HIDDEN 1024
#define BSZ    4
#define TLEN   512
#define ROWS   (BSZ*TLEN)          /* 2048 */

typedef __attribute__((ext_vector_type(16))) __bf16 v16bf;
typedef __attribute__((ext_vector_type(8)))  float  v8f;
typedef __attribute__((ext_vector_type(4)))  unsigned int u32x4;
typedef unsigned short u16;
typedef unsigned int   u32;

union FragB {
  v16bf v;
  u32x4 q[2];
  u16   u[16];
};

__device__ __forceinline__ u16 f2bf(float f) {
  union { float f; u32 u; } x; x.f = f;
  u32 u = x.u;
  u += 0x7fffu + ((u >> 16) & 1u);   // round-to-nearest-even
  return (u16)(u >> 16);
}

// ---------------------------------------------------------------------------
// 1) Embedding: x[b,t,e] = tok_emb[idx[b,t],e] + pos_emb[t,e]
// ---------------------------------------------------------------------------
__global__ void embed_kernel(const int* __restrict__ idx,
                             const float* __restrict__ tok_emb,
                             const float* __restrict__ pos_emb,
                             float* __restrict__ x) {
  int gid = blockIdx.x * blockDim.x + threadIdx.x;   // ROWS*EMBED threads
  if (gid >= ROWS * EMBED) return;
  int e  = gid & (EMBED - 1);
  int bt = gid >> 8;
  int t  = bt & (TLEN - 1);
  int tok = idx[bt];
  x[gid] = tok_emb[(size_t)tok * EMBED + e] + pos_emb[t * EMBED + e];
}

// ---------------------------------------------------------------------------
// 2) QKV projection (scalar f32; <1% of FLOPs): q[n,b,t,h] = x[b,t,:]·Wq[n,:,h]
// ---------------------------------------------------------------------------
__global__ void qkv_kernel(const float* __restrict__ x,
                           const float* __restrict__ Wq,
                           const float* __restrict__ Wk,
                           const float* __restrict__ Wv,
                           float* __restrict__ q,
                           float* __restrict__ k,
                           float* __restrict__ v) {
  int gid = blockIdx.x * blockDim.x + threadIdx.x;   // NHEAD*BSZ*TLEN*HEAD
  int h = gid & (HEAD - 1);
  int n = (gid >> 5) & (NHEAD - 1);
  int t = (gid >> 8) & (TLEN - 1);
  int b = gid >> 17;
  const float* xr = x + ((size_t)(b * TLEN + t)) * EMBED;
  const float* wqb = Wq + (size_t)n * EMBED * HEAD + h;
  const float* wkb = Wk + (size_t)n * EMBED * HEAD + h;
  const float* wvb = Wv + (size_t)n * EMBED * HEAD + h;
  float aq = 0.f, ak = 0.f, av = 0.f;
  for (int e = 0; e < EMBED; ++e) {
    float xe = xr[e];
    aq += xe * wqb[e * HEAD];
    ak += xe * wkb[e * HEAD];
    av += xe * wvb[e * HEAD];
  }
  size_t o = (((size_t)(n * BSZ + b) * TLEN + t) * HEAD) + h;
  q[o] = aq; k[o] = ak; v[o] = av;
}

// ---------------------------------------------------------------------------
// 3) Scores (stored transposed: scoresT[n][b][s][t]), causal mask t>=s,
//    scale = EMBED^-0.5 = 1/16 (reference scales by embed size).
// ---------------------------------------------------------------------------
__global__ void scores_kernel(const float* __restrict__ q,
                              const float* __restrict__ k,
                              float* __restrict__ scoresT) {
  int gid = blockIdx.x * blockDim.x + threadIdx.x;   // NHEAD*BSZ*TLEN*TLEN
  int t = gid & (TLEN - 1);
  int s = (gid >> 9) & (TLEN - 1);
  int b = (gid >> 18) & (BSZ - 1);
  int n = gid >> 20;
  float val;
  if (t >= s) {
    const float* qr = q + (((size_t)(n * BSZ + b) * TLEN + t) * HEAD);
    const float* kr = k + (((size_t)(n * BSZ + b) * TLEN + s) * HEAD);
    float acc = 0.f;
    #pragma unroll
    for (int h = 0; h < HEAD; ++h) acc += qr[h] * kr[h];
    val = acc * 0.0625f;
  } else {
    val = -__builtin_inff();
  }
  scoresT[gid] = val;   // gid == ((n*BSZ+b)*TLEN + s)*TLEN + t
}

// ---------------------------------------------------------------------------
// 4) Softmax over the QUERY axis t (contiguous row in scoresT), in place.
// ---------------------------------------------------------------------------
__global__ __launch_bounds__(256) void colsoftmax_kernel(float* __restrict__ scoresT) {
  __shared__ float red[256];
  int tid = threadIdx.x;
  float* row = scoresT + (size_t)blockIdx.x * TLEN;
  float v0 = row[tid], v1 = row[tid + 256];
  red[tid] = fmaxf(v0, v1);
  __syncthreads();
  for (int off = 128; off > 0; off >>= 1) {
    if (tid < off) red[tid] = fmaxf(red[tid], red[tid + off]);
    __syncthreads();
  }
  float m = red[0];
  __syncthreads();
  float e0 = __expf(v0 - m), e1 = __expf(v1 - m);   // exp(-inf)=0 handles mask
  red[tid] = e0 + e1;
  __syncthreads();
  for (int off = 128; off > 0; off >>= 1) {
    if (tid < off) red[tid] += red[tid + off];
    __syncthreads();
  }
  float inv = 1.f / red[0];
  row[tid] = e0 * inv;
  row[tid + 256] = e1 * inv;
}

// ---------------------------------------------------------------------------
// 5) out[b,t,n*32+h] = sum_{s<=t} wei[t,s]*v[s,h]  (wei read from weiT[s][t])
// ---------------------------------------------------------------------------
__global__ void attnout_kernel(const float* __restrict__ weiT,
                               const float* __restrict__ v,
                               float* __restrict__ attn,
                               u16* __restrict__ attnb) {
  int gid = blockIdx.x * blockDim.x + threadIdx.x;   // BSZ*TLEN*NHEAD*HEAD
  int h = gid & (HEAD - 1);
  int n = (gid >> 5) & (NHEAD - 1);
  int t = (gid >> 8) & (TLEN - 1);
  int b = gid >> 17;
  const float* wbase = weiT + ((size_t)(n * BSZ + b) * TLEN) * TLEN + t;
  const float* vbase = v + ((size_t)(n * BSZ + b) * TLEN) * HEAD + h;
  float acc = 0.f;
  for (int s = 0; s <= t; ++s)
    acc += wbase[(size_t)s * TLEN] * vbase[(size_t)s * HEAD];
  size_t o = ((size_t)(b * TLEN + t)) * EMBED + n * HEAD + h;
  attn[o] = acc;
  attnb[o] = f2bf(acc);
}

// ---------------------------------------------------------------------------
// 6) Tile-transpose + f32->bf16 convert:  BT[n,k] = bf16(B[k,n])
// ---------------------------------------------------------------------------
__global__ __launch_bounds__(256) void transpose_bf16_kernel(
    const float* __restrict__ B, u16* __restrict__ BT, int K, int N) {
  __shared__ u16 tile[64][65];
  int tid = threadIdx.x;
  int tx = tid & 63;        // fast dim
  int ty = tid >> 6;        // 0..3, each covers 16 rows
  int n0 = blockIdx.x * 64;
  int k0 = blockIdx.y * 64;
  #pragma unroll
  for (int r = 0; r < 16; ++r) {
    int kk = ty * 16 + r;
    int nn = n0 + tx;
    float f = (nn < N) ? B[(size_t)(k0 + kk) * N + nn] : 0.f;
    tile[tx][kk] = f2bf(f);
  }
  __syncthreads();
  #pragma unroll
  for (int r = 0; r < 16; ++r) {
    int nn = n0 + ty * 16 + r;
    if (nn < N) BT[(size_t)nn * K + k0 + tx] = tile[ty * 16 + r][tx];
  }
}

// ---------------------------------------------------------------------------
// 7) WMMA bf16 GEMM:  C[M,N] = A[M,K](bf16,row) * BT[N,K](bf16,row)^T + bias
//    8 waves/block; each wave: 64x32 tile = 8x v_wmma_f32_16x16x32_bf16/kstep.
//    Requires M % 256 == 0, K % 32 == 0 (true for all call sites).
//    N-edge handling: B row index is CLAMPED (no divergence in the K-loop);
//    lanes with col >= N accumulate garbage but are never stored.
// ---------------------------------------------------------------------------
__global__ __launch_bounds__(256) void gemm_bf16_kernel(
    const u16* __restrict__ A, const u16* __restrict__ BT,
    const float* __restrict__ bias,
    float* __restrict__ Cf, u16* __restrict__ Cb,
    int M, int N, int K, int relu) {
  int tid  = threadIdx.x;
  int lane = tid & 31;
  int wave = tid >> 5;
  int wm = wave >> 1;                 // 0..3
  int wn = wave & 1;                  // 0..1
  int m0 = blockIdx.y * 256 + wm * 64;
  int n0 = blockIdx.x * 64  + wn * 32;
  int lm = lane & 15;
  int lh = lane >> 4;

  // Hoisted, loop-invariant fragment base pointers (clamped for N edge).
  const u16* aptr[4];
  #pragma unroll
  for (int i = 0; i < 4; ++i)
    aptr[i] = A + (size_t)(m0 + i * 16 + lm) * K + lh * 8;
  const u16* bptr[2];
  #pragma unroll
  for (int j = 0; j < 2; ++j) {
    int nn = n0 + j * 16 + lm;
    int nc = nn < N ? nn : (N - 1);   // clamp: OOB lanes read a valid row
    bptr[j] = BT + (size_t)nc * K + lh * 8;
  }

  v8f acc[4][2];
  #pragma unroll
  for (int i = 0; i < 4; ++i)
    #pragma unroll
    for (int j = 0; j < 2; ++j)
      acc[i][j] = v8f{0.f,0.f,0.f,0.f,0.f,0.f,0.f,0.f};

  for (int k0 = 0; k0 < K; k0 += 32) {
    FragB a[4], b[2];
    // A fragments: lane holds row m, K pairs split by half-wave (ISA 16-bit A layout)
    #pragma unroll
    for (int i = 0; i < 4; ++i) {
      a[i].q[0] = *(const u32x4*)(aptr[i] + k0);
      a[i].q[1] = *(const u32x4*)(aptr[i] + k0 + 16);
    }
    // B fragments: lane holds col n, K split by half-wave (same layout as A)
    #pragma unroll
    for (int j = 0; j < 2; ++j) {
      b[j].q[0] = *(const u32x4*)(bptr[j] + k0);
      b[j].q[1] = *(const u32x4*)(bptr[j] + k0 + 16);
    }
    #pragma unroll
    for (int i = 0; i < 4; ++i)
      #pragma unroll
      for (int j = 0; j < 2; ++j)
        acc[i][j] = __builtin_amdgcn_wmma_f32_16x16x32_bf16(
            false, a[i].v, false, b[j].v, (short)0, acc[i][j], false, false);
  }

  #pragma unroll
  for (int j = 0; j < 2; ++j) {
    int col = n0 + j * 16 + lm;
    bool cok = col < N;
    int colc = cok ? col : (N - 1);
    float bv = (bias != nullptr) ? bias[colc] : 0.f;
    #pragma unroll
    for (int i = 0; i < 4; ++i) {
      #pragma unroll
      for (int r = 0; r < 8; ++r) {
        int row = m0 + i * 16 + lh * 8 + r;
        float val = acc[i][j][r] + bv;
        if (relu) val = fmaxf(val, 0.f);
        if (cok) {
          size_t o = (size_t)row * N + col;
          if (Cf) Cf[o] = val;
          if (Cb) Cb[o] = f2bf(val);
        }
      }
    }
  }
}

// ---------------------------------------------------------------------------
// 8) Per-row (vocab) online softmax stats + per-row NLL
// ---------------------------------------------------------------------------
__global__ __launch_bounds__(256) void rowstats_kernel(
    const float* __restrict__ logits, const int* __restrict__ targets,
    float* __restrict__ rowm, float* __restrict__ rows,
    float* __restrict__ lossrows) {
  __shared__ float sm[256], ss[256];
  int tid = threadIdx.x;
  int row = blockIdx.x;
  const float* lr = logits + (size_t)row * VOCAB;
  float m = -__builtin_inff(), s = 0.f;
  for (int c = tid; c < VOCAB; c += 256) {
    float l = lr[c];
    float nm = fmaxf(m, l);
    s = s * __expf(m - nm) + __expf(l - nm);
    m = nm;
  }
  sm[tid] = m; ss[tid] = s;
  __syncthreads();
  for (int off = 128; off > 0; off >>= 1) {
    if (tid < off) {
      float mo = sm[tid + off], so = ss[tid + off];
      float mm = fmaxf(sm[tid], mo);
      ss[tid] = ss[tid] * __expf(sm[tid] - mm) + so * __expf(mo - mm);
      sm[tid] = mm;
    }
    __syncthreads();
  }
  if (tid == 0) {
    float M = sm[0], S = ss[0];
    rowm[row] = M; rows[row] = S;
    float tl = lr[targets[row]];
    lossrows[row] = -(tl - M - __logf(S));
  }
}

// ---------------------------------------------------------------------------
// 9) o_prob = exp(logit - rowmax) / rowsum
// ---------------------------------------------------------------------------
__global__ void probs_kernel(const float* __restrict__ logits,
                             const float* __restrict__ rowm,
                             const float* __restrict__ rows,
                             float* __restrict__ oprob) {
  long gid = (long)blockIdx.x * blockDim.x + threadIdx.x;
  if (gid >= (long)ROWS * VOCAB) return;
  int row = (int)(gid / VOCAB);
  oprob[gid] = __expf(logits[gid] - rowm[row]) / rows[row];
}

// ---------------------------------------------------------------------------
// 10) Deterministic loss reduction (single block)
// ---------------------------------------------------------------------------
__global__ __launch_bounds__(256) void loss_kernel(const float* __restrict__ lossrows,
                                                   float* __restrict__ loss) {
  __shared__ float red[256];
  int tid = threadIdx.x;
  float s = 0.f;
  for (int r = tid; r < ROWS; r += 256) s += lossrows[r];
  red[tid] = s;
  __syncthreads();
  for (int off = 128; off > 0; off >>= 1) {
    if (tid < off) red[tid] += red[tid + off];
    __syncthreads();
  }
  if (tid == 0) loss[0] = red[0] * (1.f / (float)ROWS);
}

// ---------------------------------------------------------------------------
extern "C" void kernel_launch(void* const* d_in, const int* in_sizes, int n_in,
                              void* d_out, int out_size, void* d_ws, size_t ws_size,
                              hipStream_t stream) {
  (void)in_sizes; (void)n_in; (void)out_size; (void)ws_size;
  const int*   idx     = (const int*)d_in[0];
  const int*   targets = (const int*)d_in[1];
  const float* tok_emb = (const float*)d_in[2];
  const float* pos_emb = (const float*)d_in[3];
  const float* Wq      = (const float*)d_in[4];
  const float* Wk      = (const float*)d_in[5];
  const float* Wv      = (const float*)d_in[6];
  const float* W1      = (const float*)d_in[7];
  const float* b1      = (const float*)d_in[8];
  const float* W2      = (const float*)d_in[9];
  const float* b2      = (const float*)d_in[10];

  float* out    = (float*)d_out;
  float* logits = out;                              // [ROWS, VOCAB]
  float* loss   = out + (size_t)ROWS * VOCAB;       // [1]
  float* oprob  = loss + 1;                         // [ROWS, VOCAB]

  char* ws = (char*)d_ws;
  size_t off = 0;
  auto wsa = [&](size_t bytes) -> char* {
    char* p = ws + off;
    off += (bytes + 255) & ~(size_t)255;
    return p;
  };
  float* x        = (float*)wsa((size_t)ROWS * EMBED * 4);
  float* q        = (float*)wsa((size_t)NHEAD * BSZ * TLEN * HEAD * 4);
  float* k        = (float*)wsa((size_t)NHEAD * BSZ * TLEN * HEAD * 4);
  float* v        = (float*)wsa((size_t)NHEAD * BSZ * TLEN * HEAD * 4);
  float* scoresT  = (float*)wsa((size_t)NHEAD * BSZ * TLEN * TLEN * 4);  // 32 MB
  float* attn     = (float*)wsa((size_t)ROWS * EMBED * 4);
  u16*   attnb    = (u16*)  wsa((size_t)ROWS * EMBED * 2);
  u16*   hb       = (u16*)  wsa((size_t)ROWS * HIDDEN * 2);
  u16*   w1t      = (u16*)  wsa((size_t)HIDDEN * EMBED * 2);             // [N=1024,K=256]
  u16*   w2t      = (u16*)  wsa((size_t)VOCAB * HIDDEN * 2);             // [N=50257,K=1024] ~103MB (L2-resident)
  float* rowm     = (float*)wsa((size_t)ROWS * 4);
  float* rows     = (float*)wsa((size_t)ROWS * 4);
  float* lossrows = (float*)wsa((size_t)ROWS * 4);

  // 1) embedding
  embed_kernel<<<(ROWS * EMBED) / 256, 256, 0, stream>>>(idx, tok_emb, pos_emb, x);
  // 2) qkv
  qkv_kernel<<<(NHEAD * BSZ * TLEN * HEAD) / 256, 256, 0, stream>>>(x, Wq, Wk, Wv, q, k, v);
  // 3) scores (transposed, masked)
  scores_kernel<<<(NHEAD * BSZ * TLEN * TLEN) / 256, 256, 0, stream>>>(q, k, scoresT);
  // 4) softmax over query axis, in place
  colsoftmax_kernel<<<NHEAD * BSZ * TLEN, 256, 0, stream>>>(scoresT);
  // 5) attention output (f32 + bf16)
  attnout_kernel<<<(BSZ * TLEN * NHEAD * HEAD) / 256, 256, 0, stream>>>(scoresT, v, attn, attnb);
  // 6) weight transpose+convert (W1 -> w1t, W2 -> w2t)
  {
    dim3 g1((HIDDEN + 63) / 64, EMBED / 64);
    transpose_bf16_kernel<<<g1, 256, 0, stream>>>(W1, w1t, EMBED, HIDDEN);
    dim3 g2((VOCAB + 63) / 64, HIDDEN / 64);
    transpose_bf16_kernel<<<g2, 256, 0, stream>>>(W2, w2t, HIDDEN, VOCAB);
  }
  // 7) FFN: h = relu(attn @ W1 + b1)  -> bf16
  {
    dim3 g(HIDDEN / 64, ROWS / 256);
    gemm_bf16_kernel<<<g, 256, 0, stream>>>(attnb, w1t, b1, nullptr, hb,
                                            ROWS, HIDDEN, EMBED, 1);
  }
  // 8) LM head: logits = h @ W2 + b2  -> f32 (into d_out)
  {
    dim3 g((VOCAB + 63) / 64, ROWS / 256);
    gemm_bf16_kernel<<<g, 256, 0, stream>>>(hb, w2t, b2, logits, nullptr,
                                            ROWS, VOCAB, HIDDEN, 0);
  }
  // 9) row softmax stats + per-row loss
  rowstats_kernel<<<ROWS, 256, 0, stream>>>(logits, targets, rowm, rows, lossrows);
  // 10) probabilities
  {
    long total = (long)ROWS * VOCAB;
    int blocks = (int)((total + 255) / 256);
    probs_kernel<<<blocks, 256, 0, stream>>>(logits, rowm, rows, oprob);
  }
  // 11) loss
  loss_kernel<<<1, 256, 0, stream>>>(lossrows, loss);
}
